// MambaBlock_635655160374
// MI455X (gfx1250) — matrix operand order repaired and checked by reading, not compiled, not run on hardware
//
#include <hip/hip_runtime.h>

#define BB   8
#define LL   4096
#define DMD  384
#define DI   768
#define DSZ  16
#define DTR  24
#define HID  1536
#define NR   (BB * LL)          /* 32768 rows of tokens */

typedef unsigned int  u32;
typedef unsigned short u16;
typedef __bf16 v16bf __attribute__((ext_vector_type(16)));
typedef float  v8f   __attribute__((ext_vector_type(8)));
typedef u32    u32x4 __attribute__((ext_vector_type(4)));

union Frag { v16bf v; u32x4 q[2]; };

__device__ __forceinline__ u16 f2bf(float f) {
  u32 u = __float_as_uint(f);
  u = (u + 0x7FFFu + ((u >> 16) & 1u)) >> 16;   // round-to-nearest-even
  return (u16)u;
}
__device__ __forceinline__ float bf2f(u16 h) {
  return __uint_as_float((u32)h << 16);
}

// Low 32 bits of a generic pointer to __shared__ = LDS byte offset.
__device__ __forceinline__ u32 lds_off(const void* p) {
  return (u32)(size_t)p;
}

// gfx1250 async copy global -> LDS (16B per lane), tracked by ASYNCcnt.
__device__ __forceinline__ void async_ld_b128(u32 lds_dst, const void* gsrc) {
  asm volatile("global_load_async_to_lds_b128 %0, %1, off"
               :: "v"(lds_dst), "v"(gsrc) : "memory");
}
__device__ __forceinline__ void wait_asynccnt0() {
  asm volatile("s_wait_asynccnt 0x0" ::: "memory");
}

/* ---------------- weight prep ---------------- */

__global__ void convert_bf16_kernel(const float* __restrict__ src,
                                    u16* __restrict__ dst, int n) {
  int i = blockIdx.x * blockDim.x + threadIdx.x;
  if (i < n) dst[i] = f2bf(src[i]);
}

// copy src (rows_s x cols_s) into dst (rows_d x cols_d), zero padding
__global__ void pad_convert_kernel(const float* __restrict__ src,
                                   u16* __restrict__ dst,
                                   int rows_s, int cols_s,
                                   int rows_d, int cols_d) {
  int i = blockIdx.x * blockDim.x + threadIdx.x;
  if (i >= rows_d * cols_d) return;
  int r = i / cols_d, c = i % cols_d;
  dst[i] = (r < rows_s && c < cols_s) ? f2bf(src[r * cols_s + c]) : (u16)0;
}

__global__ void neg_exp_kernel(const float* __restrict__ src,
                               float* __restrict__ dst, int n) {
  int i = blockIdx.x * blockDim.x + threadIdx.x;
  if (i < n) dst[i] = -__expf(src[i]);
}

/* ---------------- rmsnorm (optionally fused residual add) ---------------- */
template <bool ADD>
__global__ __launch_bounds__(128)
void rmsnorm_kernel(const float* __restrict__ x, const float* __restrict__ add,
                    const float* __restrict__ w, float* __restrict__ x_out,
                    u16* __restrict__ nrm_bf) {
  __shared__ float red[4];
  size_t row = blockIdx.x;
  const float* xr = x + row * DMD;
  float v[3];
  float ss = 0.f;
#pragma unroll
  for (int i = 0; i < 3; ++i) {
    int c = threadIdx.x + i * 128;
    float t = xr[c];
    if (ADD) t += add[row * DMD + c];
    v[i] = t;
    ss += t * t;
  }
#pragma unroll
  for (int o = 16; o; o >>= 1) ss += __shfl_down(ss, o, 32);
  if ((threadIdx.x & 31) == 0) red[threadIdx.x >> 5] = ss;
  __syncthreads();
  float rs = rsqrtf((red[0] + red[1] + red[2] + red[3]) / (float)DMD + 1e-5f);
#pragma unroll
  for (int i = 0; i < 3; ++i) {
    int c = threadIdx.x + i * 128;
    if (ADD) x_out[row * DMD + c] = v[i];
    nrm_bf[row * DMD + c] = f2bf(v[i] * rs * w[c]);
  }
}

/* ---------------- bf16 WMMA GEMM with async-LDS B staging ----------------
 * C[M,N] = A[M,K] * B[N,K]^T, A/B bf16 row-major (K contiguous), fp32 acc.
 * Block: 256 threads = 8 waves (4 M x 2 N), block tile 128x64,
 * wave tile 32x32 = 2x2 v_wmma_f32_16x16x32_bf16.
 * B tile (64x32 bf16 = 4KB) staged through LDS via
 * global_load_async_to_lds_b128 (double-buffered, ASYNCcnt + barrier).
 * A fragments loaded directly from global (b128).
 * OUT_MODE: 0 = fp32 store, 1 = bf16 store, 2 = fp32 accumulate (+bias).
 * Requires M%128==0, N%64==0, K%32==0.
 */
template <int OUT_MODE>
__global__ __launch_bounds__(256)
void gemm_bf16_kernel(const u16* __restrict__ A, const u16* __restrict__ Bw,
                      void* __restrict__ Cout, const float* __restrict__ bias,
                      int M, int N, int K) {
  __shared__ u16 Btile[2][64 * 32];   // [buf][row*32 + k]
  const int tid  = threadIdx.x;
  const int lane = tid & 31;
  const int wave = tid >> 5;
  const int rowBase  = blockIdx.x * 128 + (wave & 3) * 32;
  const int colBlock = blockIdx.y * 64;
  const int colWave  = (wave >> 2) * 32;   // 0 or 32 inside the 64-wide tile
  const int ml = lane & 15;   // M index (A frag) / N index (B frag)
  const int hi = lane >> 4;   // K-half selector

  // async staging map: thread t copies 16B chunk (row = t>>2, chunk = t&3)
  const int bn = tid >> 2;
  const int bc = tid & 3;
  const u16* bSrc = Bw + (size_t)(colBlock + bn) * K + bc * 8;
  const u32 dst0 = lds_off(&Btile[0][bn * 32 + bc * 8]);
  const u32 dst1 = lds_off(&Btile[1][bn * 32 + bc * 8]);

  const u16* aPtr0 = A + (size_t)(rowBase + ml) * K;
  const u16* aPtr1 = A + (size_t)(rowBase + 16 + ml) * K;

  // prologue: stage first B tile into buffer 0
  async_ld_b128(dst0, bSrc);

  v8f acc[2][2] = {};
  int buf = 0;
  for (int k0 = 0; k0 < K; k0 += 32, buf ^= 1) {
    wait_asynccnt0();       // this wave's async copies landed in LDS
    __syncthreads();        // all waves' copies visible; prior reads done
    if (k0 + 32 < K) {      // stage next tile into the other buffer
      async_ld_b128(buf ? dst0 : dst1, bSrc + k0 + 32);
      __builtin_prefetch(aPtr0 + k0 + 32, 0, 1);  // gfx1250 global_prefetch
    }

    Frag a0, a1, b0, b1;
    // A 16x32 bf16: lane ml gets K = k0+hi*8..+7 (v0..3) and k0+16+hi*8..+7 (v4..7)
    a0.q[0] = *(const u32x4*)(aPtr0 + k0 + hi * 8);
    a0.q[1] = *(const u32x4*)(aPtr0 + k0 + 16 + hi * 8);
    a1.q[0] = *(const u32x4*)(aPtr1 + k0 + hi * 8);
    a1.q[1] = *(const u32x4*)(aPtr1 + k0 + 16 + hi * 8);
    // B 32x16 bf16 from LDS: lane n=ml reads K = hi*8..+7 (v0..3) and 16+hi*8..+7 (v4..7)
    const u16* bt = &Btile[buf][0];
    b0.q[0] = *(const u32x4*)(bt + (colWave + ml) * 32 + hi * 8);
    b0.q[1] = *(const u32x4*)(bt + (colWave + ml) * 32 + 16 + hi * 8);
    b1.q[0] = *(const u32x4*)(bt + (colWave + 16 + ml) * 32 + hi * 8);
    b1.q[1] = *(const u32x4*)(bt + (colWave + 16 + ml) * 32 + 16 + hi * 8);

    acc[0][0] = __builtin_amdgcn_wmma_f32_16x16x32_bf16(false, a0.v, false, b0.v,
                                                        (short)0, acc[0][0], false, false);
    acc[0][1] = __builtin_amdgcn_wmma_f32_16x16x32_bf16(false, a0.v, false, b1.v,
                                                        (short)0, acc[0][1], false, false);
    acc[1][0] = __builtin_amdgcn_wmma_f32_16x16x32_bf16(false, a1.v, false, b0.v,
                                                        (short)0, acc[1][0], false, false);
    acc[1][1] = __builtin_amdgcn_wmma_f32_16x16x32_bf16(false, a1.v, false, b1.v,
                                                        (short)0, acc[1][1], false, false);
  }

#pragma unroll
  for (int i = 0; i < 2; ++i) {
#pragma unroll
    for (int j = 0; j < 2; ++j) {
      int col = colBlock + colWave + j * 16 + ml;
      float bv = bias ? bias[col] : 0.f;
#pragma unroll
      for (int r = 0; r < 8; ++r) {
        int row = rowBase + i * 16 + hi * 8 + r;   // C layout: v_r = M=r(+8 for hi half)
        size_t off = (size_t)row * N + col;
        float val = acc[i][j][r] + bv;
        if (OUT_MODE == 0)      ((float*)Cout)[off] = val;
        else if (OUT_MODE == 1) ((u16*)Cout)[off] = f2bf(val);
        else                    ((float*)Cout)[off] += val;
      }
    }
  }
}

/* ---------------- depthwise causal conv (width 4) + SiLU ---------------- */
__global__ void conv_silu_kernel(const u16* __restrict__ xz,
                                 const float* __restrict__ cw,
                                 const float* __restrict__ cb,
                                 u16* __restrict__ u_out) {
  size_t idx = (size_t)blockIdx.x * blockDim.x + threadIdx.x;
  if (idx >= (size_t)NR * DI) return;
  int d = (int)(idx % DI);
  size_t bl = idx / DI;
  int l = (int)(bl % LL);
  size_t b = bl / LL;
  float acc = cb[d];
#pragma unroll
  for (int j = 0; j < 4; ++j) {
    int ls = l - 3 + j;
    if (ls >= 0)
      acc += cw[d * 4 + j] * bf2f(xz[(b * LL + ls) * (size_t)(2 * DI) + d]);
  }
  float s = acc / (1.f + __expf(-acc));
  u_out[idx] = f2bf(s);
}

/* ---------------- gather dt columns into padded bf16 (K=32) ---------------- */
__global__ void copy_dt_kernel(const float* __restrict__ xdbl,
                               u16* __restrict__ dt_bf) {
  int i = blockIdx.x * blockDim.x + threadIdx.x;
  if (i >= NR * 32) return;
  int m = i >> 5, r = i & 31;
  dt_bf[i] = (r < DTR) ? f2bf(xdbl[(size_t)m * 64 + r]) : (u16)0;
}

/* ---------------- selective scan ----------------
 * 48 blocks x 128 threads: block = (batch b, 128-wide d chunk).
 * Per thread: h[16], A[d][16] in registers; B_t/C_t broadcast via LDS.
 */
__global__ __launch_bounds__(128)
void scan_kernel(const float* __restrict__ dlin, const float* __restrict__ b_dt,
                 const u16* __restrict__ u_bf, const float* __restrict__ xdbl,
                 const float* __restrict__ Aneg, float* __restrict__ y) {
  __shared__ float BC[2][32];
  int b = blockIdx.x / (DI / 128);
  int d = (blockIdx.x % (DI / 128)) * 128 + threadIdx.x;
  float h[DSZ], Ad[DSZ];
#pragma unroll
  for (int s = 0; s < DSZ; ++s) { h[s] = 0.f; Ad[s] = Aneg[d * DSZ + s]; }
  float bd = b_dt[d];
  for (int l = 0; l < LL; ++l) {
    int buf = l & 1;
    if (threadIdx.x < 32)
      BC[buf][threadIdx.x] = xdbl[((size_t)b * LL + l) * 64 + DTR + threadIdx.x];
    __syncthreads();
    size_t off = ((size_t)b * LL + l) * DI + d;
    float xv = dlin[off] + bd;
    float delta = (xv > 20.f) ? xv : log1pf(__expf(xv));   // softplus
    float du = delta * bf2f(u_bf[off]);
    float yt = 0.f;
#pragma unroll
    for (int s = 0; s < DSZ; ++s) {
      h[s] = __expf(delta * Ad[s]) * h[s] + du * BC[buf][s];
      yt += h[s] * BC[buf][16 + s];
    }
    y[off] = yt;
  }
}

/* ---------------- y = (scan + u*D) * silu(z) -> bf16 ---------------- */
__global__ void gate_kernel(const float* __restrict__ y, const u16* __restrict__ u_bf,
                            const u16* __restrict__ xz, const float* __restrict__ Dv,
                            u16* __restrict__ ybf) {
  size_t idx = (size_t)blockIdx.x * blockDim.x + threadIdx.x;
  if (idx >= (size_t)NR * DI) return;
  int d = (int)(idx % DI);
  size_t row = idx / DI;
  float z = bf2f(xz[row * (size_t)(2 * DI) + DI + d]);
  float val = (y[idx] + bf2f(u_bf[idx]) * Dv[d]) * (z / (1.f + __expf(-z)));
  ybf[idx] = f2bf(val);
}

/* ---------------- exact GELU on bf16 ---------------- */
__global__ void gelu_kernel(const u16* __restrict__ hin, const float* __restrict__ bias,
                            u16* __restrict__ hout) {
  size_t idx = (size_t)blockIdx.x * blockDim.x + threadIdx.x;
  if (idx >= (size_t)NR * HID) return;
  float xv = bf2f(hin[idx]) + bias[idx % HID];
  float g = 0.5f * xv * (1.f + erff(xv * 0.70710678118654752f));
  hout[idx] = f2bf(g);
}

/* ================= launch ================= */
extern "C" void kernel_launch(void* const* d_in, const int* in_sizes, int n_in,
                              void* d_out, int out_size, void* d_ws, size_t ws_size,
                              hipStream_t stream) {
  const float* x       = (const float*)d_in[0];
  const float* norm1_w = (const float*)d_in[1];
  const float* w_in    = (const float*)d_in[2];
  const float* conv_w  = (const float*)d_in[3];
  const float* conv_b  = (const float*)d_in[4];
  const float* w_xproj = (const float*)d_in[5];
  const float* w_dt    = (const float*)d_in[6];
  const float* b_dt    = (const float*)d_in[7];
  const float* A_log   = (const float*)d_in[8];
  const float* Dvec    = (const float*)d_in[9];
  const float* w_out   = (const float*)d_in[10];
  const float* norm2_w = (const float*)d_in[11];
  const float* fc1_w   = (const float*)d_in[12];
  const float* fc1_b   = (const float*)d_in[13];
  const float* fc2_w   = (const float*)d_in[14];
  const float* fc2_b   = (const float*)d_in[15];
  float* out = (float*)d_out;

  char* ws = (char*)d_ws;
  size_t ofs = 0;
  auto alloc = [&](size_t bytes) -> char* {
    char* p = ws + ofs;
    ofs += (bytes + 255) & ~(size_t)255;
    return p;
  };
  u16*   w_in_bf  = (u16*)alloc((size_t)2 * DI * DMD * 2);      // 1536x384
  u16*   w_xp_bf  = (u16*)alloc((size_t)64 * DI * 2);           // 64x768 (padded)
  u16*   w_dt_bf  = (u16*)alloc((size_t)DI * 32 * 2);           // 768x32 (padded)
  u16*   w_out_bf = (u16*)alloc((size_t)DMD * DI * 2);          // 384x768
  u16*   fc1_bf   = (u16*)alloc((size_t)HID * DMD * 2);         // 1536x384
  u16*   fc2_bf   = (u16*)alloc((size_t)DMD * HID * 2);         // 384x1536
  float* Aneg     = (float*)alloc((size_t)DI * DSZ * 4);
  u16*   xn_bf    = (u16*)alloc((size_t)NR * DMD * 2);          // norm1/norm2 out
  u16*   xz_bf    = (u16*)alloc((size_t)NR * 2 * DI * 2);       // in_proj out
  u16*   u_bf     = (u16*)alloc((size_t)NR * DI * 2);           // conv+silu out
  float* xdbl     = (float*)alloc((size_t)NR * 64 * 4);         // x_proj out (N pad 64)
  u16*   dt_bf    = (u16*)alloc((size_t)NR * 32 * 2);
  float* dlin     = (float*)alloc((size_t)NR * DI * 4);         // reused: h1 bf16 (NR*HID)
  float* yscan    = (float*)alloc((size_t)NR * DI * 4);         // reused: gelu bf16 (NR*HID)
  u16*   ybf      = (u16*)alloc((size_t)NR * DI * 2);
  float* mamba_o  = (float*)alloc((size_t)NR * DMD * 4);
  (void)ws_size; (void)in_sizes; (void)n_in; (void)out_size;

  const int T = 256;
  // weight prep (deterministic each call)
  convert_bf16_kernel<<<(2 * DI * DMD + T - 1) / T, T, 0, stream>>>(w_in, w_in_bf, 2 * DI * DMD);
  convert_bf16_kernel<<<(DMD * DI + T - 1) / T, T, 0, stream>>>(w_out, w_out_bf, DMD * DI);
  convert_bf16_kernel<<<(HID * DMD + T - 1) / T, T, 0, stream>>>(fc1_w, fc1_bf, HID * DMD);
  convert_bf16_kernel<<<(DMD * HID + T - 1) / T, T, 0, stream>>>(fc2_w, fc2_bf, DMD * HID);
  pad_convert_kernel<<<(64 * DI + T - 1) / T, T, 0, stream>>>(w_xproj, w_xp_bf,
                                                              DTR + 2 * DSZ, DI, 64, DI);
  pad_convert_kernel<<<(DI * 32 + T - 1) / T, T, 0, stream>>>(w_dt, w_dt_bf, DI, DTR, DI, 32);
  neg_exp_kernel<<<(DI * DSZ + T - 1) / T, T, 0, stream>>>(A_log, Aneg, DI * DSZ);

  // 1) rmsnorm1
  rmsnorm_kernel<false><<<NR, 128, 0, stream>>>(x, nullptr, norm1_w, nullptr, xn_bf);
  // 2) in_proj: (NR x 384) x (1536 x 384)^T -> xz bf16
  gemm_bf16_kernel<1><<<dim3(NR / 128, (2 * DI) / 64), 256, 0, stream>>>(
      xn_bf, w_in_bf, xz_bf, nullptr, NR, 2 * DI, DMD);
  // 3) depthwise conv + silu
  conv_silu_kernel<<<((size_t)NR * DI + T - 1) / T, T, 0, stream>>>(xz_bf, conv_w, conv_b, u_bf);
  // 4) x_proj: (NR x 768) x (64 x 768)^T -> xdbl fp32 (cols 0..55 valid)
  gemm_bf16_kernel<0><<<dim3(NR / 128, 64 / 64), 256, 0, stream>>>(
      u_bf, w_xp_bf, xdbl, nullptr, NR, 64, DI);
  // 5) dt pad -> bf16, delta_lin = dt x w_dt^T
  copy_dt_kernel<<<(NR * 32 + T - 1) / T, T, 0, stream>>>(xdbl, dt_bf);
  gemm_bf16_kernel<0><<<dim3(NR / 128, DI / 64), 256, 0, stream>>>(
      dt_bf, w_dt_bf, dlin, nullptr, NR, DI, 32);
  // 6) selective scan
  scan_kernel<<<BB * (DI / 128), 128, 0, stream>>>(dlin, b_dt, u_bf, xdbl, Aneg, yscan);
  // 7) gating
  gate_kernel<<<((size_t)NR * DI + T - 1) / T, T, 0, stream>>>(yscan, u_bf, xz_bf, Dvec, ybf);
  // 8) out_proj -> mamba_o fp32
  gemm_bf16_kernel<0><<<dim3(NR / 128, DMD / 64), 256, 0, stream>>>(
      ybf, w_out_bf, mamba_o, nullptr, NR, DMD, DI);
  // 9) residual + rmsnorm2: out = x + mamba_o ; xn_bf = norm(out)*w2
  rmsnorm_kernel<true><<<NR, 128, 0, stream>>>(x, mamba_o, norm2_w, out, xn_bf);
  // 10) fc1 -> bf16 (into dlin region), gelu -> bf16 (into yscan region)
  gemm_bf16_kernel<1><<<dim3(NR / 128, HID / 64), 256, 0, stream>>>(
      xn_bf, fc1_bf, (u16*)dlin, nullptr, NR, HID, DMD);
  gelu_kernel<<<((size_t)NR * HID + T - 1) / T, T, 0, stream>>>((const u16*)dlin, fc1_b, (u16*)yscan);
  // 11) fc2 with fused residual accumulate: out += acc + fc2_b
  gemm_bf16_kernel<2><<<dim3(NR / 128, DMD / 64), 256, 0, stream>>>(
      (const u16*)yscan, fc2_bf, out, fc2_b, NR, DMD, HID);
}
